// GraphSAGE_89309549953641
// MI455X (gfx1250) — matrix operand order repaired and checked by reading, not compiled, not run on hardware
//
#include <hip/hip_runtime.h>

#define IN_DIM 512
#define HID_DIM 256
#define OUT_DIM 128

typedef __bf16 v16bf __attribute__((ext_vector_type(16)));
typedef float  v8f   __attribute__((ext_vector_type(8)));
typedef unsigned int v4u __attribute__((ext_vector_type(4)));
typedef int v8i __attribute__((ext_vector_type(8)));
typedef int v4i __attribute__((ext_vector_type(4)));

__device__ __forceinline__ unsigned short f2bf(float f) {
  union { float f; unsigned int u; } c; c.f = f;
  unsigned int u = c.u;
  u = u + 0x7FFFu + ((u >> 16) & 1u);   // round-to-nearest-even
  return (unsigned short)(u >> 16);
}

// ---------------- degree count / reciprocal ----------------
__global__ void count_kernel(const int* __restrict__ dst, float* __restrict__ cnt, int E) {
  int e = blockIdx.x * blockDim.x + threadIdx.x;
  if (e < E) atomicAdd(&cnt[dst[e]], 1.0f);
}

__global__ void inv_kernel(float* cnt, int n) {
  int i = blockIdx.x * blockDim.x + threadIdx.x;
  if (i < n) cnt[i] = 1.0f / fmaxf(cnt[i], 1.0f);
}

// ---------------- edge scatter-add (mean numerator) ----------------
__global__ void scatter_kernel(const float* __restrict__ feat, const int* __restrict__ src,
                               const int* __restrict__ dst, float* __restrict__ agg,
                               int E, int D) {
  long long tid = (long long)blockIdx.x * blockDim.x + threadIdx.x;
  const int quads = D >> 2;
  long long total = (long long)E * quads;
  if (tid >= total) return;
  int e  = (int)(tid / quads);
  int f4 = (int)(tid % quads) << 2;
  int s = src[e], d = dst[e];
  const float4 v = *reinterpret_cast<const float4*>(feat + (size_t)s * D + f4);
  float* a = agg + (size_t)d * D + f4;
  atomicAdd(a + 0, v.x);
  atomicAdd(a + 1, v.y);
  atomicAdd(a + 2, v.z);
  atomicAdd(a + 3, v.w);
}

// ---------------- weight convert + transpose: Wt[n][k] = bf16(W[k][n]) ----------------
__global__ void wtrans_kernel(const float* __restrict__ W, unsigned short* __restrict__ Wt,
                              int K, int N) {
  int idx = blockIdx.x * blockDim.x + threadIdx.x;
  if (idx < K * N) {
    int k = idx / N, nn = idx % N;
    Wt[(size_t)nn * K + k] = f2bf(W[idx]);
  }
}

// ---------------- fused SAGE GEMM: out = act( (agg@Wl)*inv_cnt + root@Wr + b ) ----------------
// Block = 128 threads = 4 waves. Block tile: M=32, N=64, K-step 64.
// Each wave owns one 16-col slice and 2 M-tiles -> 4 v_wmma per staging round.
// B tiles (bf16, pre-transposed [N][K]) are DMA'd into LDS by the Tensor Data Mover with
// pad_enable producing a 144B LDS row stride; A tiles staged f32->bf16 by VALU.
template<int KD, int ND, bool RELU>
__global__ __launch_bounds__(128) void sage_gemm(
    const float* __restrict__ Aagg, const float* __restrict__ Aroot,
    const unsigned short* __restrict__ Wlt, const unsigned short* __restrict__ Wrt,
    const float* __restrict__ bias, const float* __restrict__ inv_cnt,
    float* __restrict__ out, int n)
{
  constexpr int SA = 72;   // LDS row stride (bf16) for A: 144B rows, 16B-aligned fragments
  constexpr int SB = 72;   // LDS row stride (bf16) for B: matches TDM pad (32 DW + 4 DW pad)
  __shared__ __align__(16) unsigned short Alds[32 * SA];
  __shared__ __align__(16) unsigned short Blds[64 * SB];

  const int t    = threadIdx.x;
  const int lane = t & 31;
  const int wave = t >> 5;
  const int ml   = lane & 15;   // M row (A) / N col (B) within 16
  const int hi   = lane >> 4;   // lane-half selector

  const int m_base = blockIdx.x * 32;
  const int col_b  = blockIdx.y * 64;

  v8f acc[2];
  acc[0] = (v8f){0.f,0.f,0.f,0.f,0.f,0.f,0.f,0.f};
  acc[1] = (v8f){0.f,0.f,0.f,0.f,0.f,0.f,0.f,0.f};

  const unsigned lds_b_base = (unsigned)(size_t)&Blds[0];

  for (int phase = 0; phase < 2; ++phase) {
    const float* A = (phase == 0) ? Aagg : Aroot;
    const unsigned short* Wt = (phase == 0) ? Wlt : Wrt;

    for (int k0 = 0; k0 < KD; k0 += 64) {
      __syncthreads();

      // --- TDM: B tile = Wt[col_b .. col_b+63][k0 .. k0+63] -> Blds (wave 0 issues) ---
      if (wave == 0) {
        unsigned long long ga =
            (unsigned long long)(size_t)(Wt + (size_t)col_b * KD + k0);
        v4u g0;
        g0.x = 1u;                                    // count=1, user D#, no gather
        g0.y = lds_b_base;                            // lds_addr (bytes)
        g0.z = (unsigned)(ga & 0xFFFFFFFFull);        // global_addr[31:0]
        g0.w = (unsigned)((ga >> 32) & 0x1FFFFFFull)  // global_addr[56:32]
             | (2u << 30);                            // type=2 (image)
        v8i g1;
        g1[0] = (1 << 16)      // data_size = 2 bytes
              | (1 << 20)      // pad_enable
              | (4 << 22)      // pad_interval: 32 DWORDs (=64 bf16 row)
              | (3 << 25);     // pad_amount: 4 DWORDs (=16B -> 144B row stride)
        g1[1] = (KD & 0xFFFF) << 16;                       // tensor_dim0 lo16
        g1[2] = ((KD >> 16) & 0xFFFF) | ((ND & 0xFFFF) << 16); // dim0 hi / dim1 lo
        g1[3] = ((ND >> 16) & 0xFFFF) | (64 << 16);        // dim1 hi / tile_dim0=64
        g1[4] = 64;                                        // tile_dim1=64, tile_dim2=0
        g1[5] = KD;                                        // tensor_dim0_stride lo32
        g1[6] = 0;
        g1[7] = 0;
        v4i gz  = (v4i){0, 0, 0, 0};
        v8i gz8 = (v8i){0, 0, 0, 0, 0, 0, 0, 0};
        __builtin_amdgcn_tensor_load_to_lds(g0, g1, gz, gz, gz8, 0);
      }

      // --- stage A tile: 32 rows x 64 K, f32 -> bf16 (all 128 threads) ---
      #pragma unroll
      for (int i = 0; i < 16; ++i) {
        int idx = i * 128 + t;              // 0..2047
        int r = idx >> 6, k = idx & 63;
        int rg = m_base + r; rg = rg < n ? rg : n - 1;
        Alds[r * SA + k] = f2bf(A[(size_t)rg * KD + k0 + k]);
      }

      if (wave == 0) __builtin_amdgcn_s_wait_tensorcnt(0);
      __syncthreads();

      // --- fragments (ISA 16-bit WMMA layouts) + 4 chained WMMAs ---
      union Frag { v16bf v; uint4 q[2]; };
      Frag b0, b1, a00, a01, a10, a11;
      const int bc = wave * 16 + ml;
      b0.q[0]  = *reinterpret_cast<const uint4*>(&Blds[bc * SB + 8 * hi]);
      b0.q[1]  = *reinterpret_cast<const uint4*>(&Blds[bc * SB + 16 + 8 * hi]);
      b1.q[0]  = *reinterpret_cast<const uint4*>(&Blds[bc * SB + 32 + 8 * hi]);
      b1.q[1]  = *reinterpret_cast<const uint4*>(&Blds[bc * SB + 48 + 8 * hi]);
      a00.q[0] = *reinterpret_cast<const uint4*>(&Alds[ml * SA + 8 * hi]);
      a00.q[1] = *reinterpret_cast<const uint4*>(&Alds[ml * SA + 16 + 8 * hi]);
      a01.q[0] = *reinterpret_cast<const uint4*>(&Alds[ml * SA + 32 + 8 * hi]);
      a01.q[1] = *reinterpret_cast<const uint4*>(&Alds[ml * SA + 48 + 8 * hi]);
      a10.q[0] = *reinterpret_cast<const uint4*>(&Alds[(16 + ml) * SA + 8 * hi]);
      a10.q[1] = *reinterpret_cast<const uint4*>(&Alds[(16 + ml) * SA + 16 + 8 * hi]);
      a11.q[0] = *reinterpret_cast<const uint4*>(&Alds[(16 + ml) * SA + 32 + 8 * hi]);
      a11.q[1] = *reinterpret_cast<const uint4*>(&Alds[(16 + ml) * SA + 48 + 8 * hi]);

      acc[0] = __builtin_amdgcn_wmma_f32_16x16x32_bf16(false, a00.v, false, b0.v,
                                                       (short)0, acc[0], false, false);
      acc[0] = __builtin_amdgcn_wmma_f32_16x16x32_bf16(false, a01.v, false, b1.v,
                                                       (short)0, acc[0], false, false);
      acc[1] = __builtin_amdgcn_wmma_f32_16x16x32_bf16(false, a10.v, false, b0.v,
                                                       (short)0, acc[1], false, false);
      acc[1] = __builtin_amdgcn_wmma_f32_16x16x32_bf16(false, a11.v, false, b1.v,
                                                       (short)0, acc[1], false, false);
    }

    if (phase == 0) {
      // fold the segment-mean divide into the accumulators: (sum @ W) * (1/cnt[row])
      #pragma unroll
      for (int mt = 0; mt < 2; ++mt) {
        #pragma unroll
        for (int r = 0; r < 8; ++r) {
          int rg = m_base + mt * 16 + r + hi * 8; rg = rg < n ? rg : n - 1;
          acc[mt][r] *= inv_cnt[rg];
        }
      }
    }
  }

  const int col = col_b + wave * 16 + ml;
  const float bv = bias[col];
  #pragma unroll
  for (int mt = 0; mt < 2; ++mt) {
    #pragma unroll
    for (int r = 0; r < 8; ++r) {
      int rg = m_base + mt * 16 + r + hi * 8;
      if (rg < n) {
        float v = acc[mt][r] + bv;
        if (RELU) v = fmaxf(v, 0.0f);
        out[(size_t)rg * ND + col] = v;
      }
    }
  }
}

extern "C" void kernel_launch(void* const* d_in, const int* in_sizes, int n_in,
                              void* d_out, int out_size, void* d_ws, size_t ws_size,
                              hipStream_t stream) {
  (void)n_in; (void)out_size; (void)ws_size;
  const float* x   = (const float*)d_in[0];
  const int*  eidx = (const int*)d_in[1];
  const float* Wl1 = (const float*)d_in[2];
  const float* bl1 = (const float*)d_in[3];
  const float* Wr1 = (const float*)d_in[4];
  const float* Wl2 = (const float*)d_in[5];
  const float* bl2 = (const float*)d_in[6];
  const float* Wr2 = (const float*)d_in[7];
  float* out = (float*)d_out;

  const int n = in_sizes[0] / IN_DIM;
  const int E = in_sizes[1] / 2;
  const int* src  = eidx;       // edge_index[0]
  const int* dstv = eidx + E;   // edge_index[1]

  // workspace carve-out: agg (reused for both layers) | h | cnt | transposed bf16 weights
  char* ws = (char*)d_ws;
  const size_t aggBytes = (size_t)n * IN_DIM * sizeof(float);
  const size_t hBytes   = (size_t)n * HID_DIM * sizeof(float);
  const size_t cntBytes = (((size_t)n * sizeof(float)) + 255u) & ~(size_t)255u;
  float* agg = (float*)ws;
  float* h   = (float*)(ws + aggBytes);
  float* cnt = (float*)(ws + aggBytes + hBytes);
  unsigned short* Wl1t = (unsigned short*)(ws + aggBytes + hBytes + cntBytes);
  unsigned short* Wr1t = Wl1t + (size_t)IN_DIM * HID_DIM;
  unsigned short* Wl2t = Wr1t + (size_t)IN_DIM * HID_DIM;
  unsigned short* Wr2t = Wl2t + (size_t)HID_DIM * OUT_DIM;

  // pre-convert + transpose weights to bf16 [N][K] (tiny, enables TDM B-tile DMA)
  wtrans_kernel<<<(IN_DIM * HID_DIM + 255) / 256, 256, 0, stream>>>(Wl1, Wl1t, IN_DIM, HID_DIM);
  wtrans_kernel<<<(IN_DIM * HID_DIM + 255) / 256, 256, 0, stream>>>(Wr1, Wr1t, IN_DIM, HID_DIM);
  wtrans_kernel<<<(HID_DIM * OUT_DIM + 255) / 256, 256, 0, stream>>>(Wl2, Wl2t, HID_DIM, OUT_DIM);
  wtrans_kernel<<<(HID_DIM * OUT_DIM + 255) / 256, 256, 0, stream>>>(Wr2, Wr2t, HID_DIM, OUT_DIM);

  // degree counts -> reciprocal (shared by both layers)
  (void)hipMemsetAsync(cnt, 0, (size_t)n * sizeof(float), stream);
  count_kernel<<<(E + 255) / 256, 256, 0, stream>>>(dstv, cnt, E);
  inv_kernel<<<(n + 255) / 256, 256, 0, stream>>>(cnt, n);

  // layer 1: aggregate x, then fused WMMA GEMM (+relu) -> h
  (void)hipMemsetAsync(agg, 0, aggBytes, stream);
  {
    long long total = (long long)E * (IN_DIM / 4);
    scatter_kernel<<<(int)((total + 255) / 256), 256, 0, stream>>>(x, src, dstv, agg, E, IN_DIM);
  }
  dim3 g1((unsigned)((n + 31) / 32), HID_DIM / 64);
  sage_gemm<IN_DIM, HID_DIM, true><<<g1, 128, 0, stream>>>(agg, x, Wl1t, Wr1t, bl1, cnt, h, n);

  // layer 2: aggregate h (reuse agg buffer), fused WMMA GEMM -> out
  (void)hipMemsetAsync(agg, 0, (size_t)n * HID_DIM * sizeof(float), stream);
  {
    long long total = (long long)E * (HID_DIM / 4);
    scatter_kernel<<<(int)((total + 255) / 256), 256, 0, stream>>>(h, src, dstv, agg, E, HID_DIM);
  }
  dim3 g2((unsigned)((n + 31) / 32), OUT_DIM / 64);
  sage_gemm<HID_DIM, OUT_DIM, false><<<g2, 128, 0, stream>>>(agg, h, Wl2t, Wr2t, bl2, cnt, out, n);
}
